// CoarseMatching_68247030333688
// MI455X (gfx1250) — matrix operand order repaired and checked by reading, not compiled, not run on hardware
//
#include <hip/hip_runtime.h>
#include <stddef.h>

// Problem constants (match reference)
#define NB      2
#define LL      4800
#define SS      4800
#define DD      256
#define H0C     60
#define W0C     80
#define BORDER  2
#define THRV    0.2f
#define INV_TEMP 10.0f          // 1/0.1
#define INV_SQRT_D 0.0625f      // 1/sqrt(256)
#define NLROWS  (NB*LL)         // 9600
#define NSROWS  (NB*SS)         // 9600
#define TOTROWS (NLROWS+NSROWS) // 19200
#define SEGS    25
#define SEGROWS (LL/SEGS)       // 192

typedef __attribute__((ext_vector_type(16))) __bf16          bf16x16;
typedef __attribute__((ext_vector_type(16))) unsigned short  u16x16;
typedef __attribute__((ext_vector_type(8)))  float           f32x8;

__device__ __forceinline__ unsigned short f32_to_bf16(float f) {
    unsigned int u = __float_as_uint(f);
    unsigned int r = (u + 0x7fffu + ((u >> 16) & 1u)) >> 16;   // RNE
    return (unsigned short)r;
}

__device__ __forceinline__ bool border_ok(int i) {
    int hh = i / W0C, ww = i % W0C;
    return (hh >= BORDER) && (hh < H0C - BORDER) && (ww >= BORDER) && (ww < W0C - BORDER);
}

// ---------------------------------------------------------------------------
// 1) Projection: f = (feat @ W^T + b) * 1/sqrt(D), output bf16 rows.
//    Virtual row space: [0, N*L) = feat_c0 flat, [N*L, N*L+N*S) = feat_c1 flat.
//    64-row register-blocked tile; W row cached per thread, feat tile in LDS.
// ---------------------------------------------------------------------------
__global__ void proj_kernel(const float* __restrict__ feat0,
                            const float* __restrict__ feat1,
                            const float* __restrict__ Wm,
                            const float* __restrict__ bias,
                            unsigned short* __restrict__ fb) {
    __shared__ __align__(16) float sf[64 * DD];   // 64 KB
    const int t  = threadIdx.x;
    const int R0 = blockIdx.x * 64;

    // stage 64 x 256 f32 tile (float4 granular, coalesced)
    for (int i = t; i < 64 * (DD / 4); i += 256) {
        int r = i >> 6;            // row within tile
        int q = i & 63;            // float4 index within row
        int grow = R0 + r;
        const float* src = (grow < NLROWS)
            ? (feat0 + (size_t)grow * DD)
            : (feat1 + (size_t)(grow - NLROWS) * DD);
        ((float4*)sf)[r * (DD / 4) + q] = ((const float4*)src)[q];
    }
    __syncthreads();

    const int j = t;  // output column owned by this thread
    float acc[64];
#pragma unroll
    for (int r = 0; r < 64; ++r) acc[r] = 0.f;

    const float4* Wv = (const float4*)(Wm + (size_t)j * DD);
    const float4* SF = (const float4*)sf;
    for (int k4 = 0; k4 < DD / 4; ++k4) {
        float4 w = Wv[k4];
#pragma unroll
        for (int r = 0; r < 64; ++r) {
            float4 f = SF[r * (DD / 4) + k4];
            acc[r] += f.x * w.x + f.y * w.y + f.z * w.z + f.w * w.w;
        }
    }
    const float bj = bias[j];
#pragma unroll
    for (int r = 0; r < 64; ++r) {
        float v = (acc[r] + bj) * INV_SQRT_D;
        fb[(size_t)(R0 + r) * DD + j] = f32_to_bf16(v);
    }
}

// ---------------------------------------------------------------------------
// 2) sim[n,l,s] = (f0[n,l,:] . f1[n,s,:]) / TEMP via bf16 WMMA.
//    64x64 block tile, 4 waves (128 threads); each wave computes a 32x32
//    register tile = 2 A frags x 2 B frags -> 4 WMMAs per K-step with full
//    fragment reuse (2 ds_load_b128 per WMMA instead of 3).
// ---------------------------------------------------------------------------
__device__ __forceinline__ bf16x16 load_a_frag(const unsigned short* tile,
                                               int rowbase, int half, int idx) {
    // ISA 16-bit A 16x32: lane m<16 holds K 0-7 / 16-23; lane>=16 holds K 8-15 / 24-31
    u16x16 u;
    const unsigned short* p = tile + (rowbase + idx) * 32;
#pragma unroll
    for (int i = 0; i < 8; ++i) {
        u[i]     = p[half * 8 + i];
        u[8 + i] = p[16 + half * 8 + i];
    }
    return __builtin_bit_cast(bf16x16, u);
}

__device__ __forceinline__ bf16x16 load_b_frag(const unsigned short* tile,
                                               int colbase, int half, int idx) {
    // B 32x16: lane<16 holds K 0-7 / 16-23, lane>=16 holds K 8-15 / 24-31 of column idx
    u16x16 u;
    const unsigned short* p = tile + (colbase + idx) * 32;
#pragma unroll
    for (int i = 0; i < 8; ++i) {
        u[i]     = p[half * 8 + i];
        u[8 + i] = p[16 + half * 8 + i];
    }
    return __builtin_bit_cast(bf16x16, u);
}

__global__ void sim_gemm_kernel(const unsigned short* __restrict__ fb,
                                float* __restrict__ sim) {
    __shared__ __align__(16) unsigned short As[64 * 32];
    __shared__ __align__(16) unsigned short Bs[64 * 32];

    const int t  = threadIdx.x;          // 0..127 (4 waves)
    const int n  = blockIdx.z;
    const int l0 = blockIdx.y * 64;
    const int s0 = blockIdx.x * 64;

    const unsigned short* A = fb + ((size_t)n * LL) * DD;                   // f0 batch n
    const unsigned short* B = fb + ((size_t)NLROWS + (size_t)n * SS) * DD;  // f1 batch n

    const int wave = t >> 5, lane = t & 31;
    const int wr = wave & 1;             // row strip of 32
    const int wc = wave >> 1;            // col strip of 32
    const int half = lane >> 4, idx = lane & 15;

    f32x8 acc00 = {}, acc01 = {}, acc10 = {}, acc11 = {};

    for (int k0 = 0; k0 < DD; k0 += 32) {
        // stage A/B K-slices into LDS: 2 x 16B per tile per thread, coalesced
#pragma unroll
        for (int i = 0; i < 2; ++i) {
            int e = t + i * 128;         // 0..255
            int r = e >> 2, q = e & 3;
            *(uint4*)(As + r * 32 + q * 8) =
                *(const uint4*)(A + (size_t)(l0 + r) * DD + k0 + q * 8);
            *(uint4*)(Bs + r * 32 + q * 8) =
                *(const uint4*)(B + (size_t)(s0 + r) * DD + k0 + q * 8);
            if (k0 + 32 < DD) {          // hint next K slice (global_prefetch_b8)
                __builtin_prefetch(A + (size_t)(l0 + r) * DD + k0 + 32 + q * 8, 0, 1);
                __builtin_prefetch(B + (size_t)(s0 + r) * DD + k0 + 32 + q * 8, 0, 1);
            }
        }
        __syncthreads();

        bf16x16 a0 = load_a_frag(As, wr * 32 + 0,  half, idx);
        bf16x16 a1 = load_a_frag(As, wr * 32 + 16, half, idx);
        bf16x16 b0 = load_b_frag(Bs, wc * 32 + 0,  half, idx);
        bf16x16 b1 = load_b_frag(Bs, wc * 32 + 16, half, idx);

        acc00 = __builtin_amdgcn_wmma_f32_16x16x32_bf16(false, a0, false, b0,
                                                        (short)0, acc00, false, false);
        acc01 = __builtin_amdgcn_wmma_f32_16x16x32_bf16(false, a0, false, b1,
                                                        (short)0, acc01, false, false);
        acc10 = __builtin_amdgcn_wmma_f32_16x16x32_bf16(false, a1, false, b0,
                                                        (short)0, acc10, false, false);
        acc11 = __builtin_amdgcn_wmma_f32_16x16x32_bf16(false, a1, false, b1,
                                                        (short)0, acc11, false, false);
        __syncthreads();
    }

    // C/D layout: VGPR v -> M = v + (half?8:0), N = idx (per 16x16 tile)
    float* simn = sim + (size_t)n * LL * SS;
    const int r0 = l0 + wr * 32 + half * 8;        // M tile 0 rows
    const int r1 = r0 + 16;                        // M tile 1 rows
    const int c0 = s0 + wc * 32 + idx;             // N tile 0 cols
    const int c1 = c0 + 16;                        // N tile 1 cols
#pragma unroll
    for (int v = 0; v < 8; ++v) {
        size_t o0 = (size_t)(r0 + v) * SS;
        size_t o1 = (size_t)(r1 + v) * SS;
        simn[o0 + c0] = acc00[v] * INV_TEMP;
        simn[o0 + c1] = acc01[v] * INV_TEMP;
        simn[o1 + c0] = acc10[v] * INV_TEMP;
        simn[o1 + c1] = acc11[v] * INV_TEMP;
    }
}

// ---------------------------------------------------------------------------
// 3a) Row stats: one block per row, online (max, sumexp), LDS tree combine.
// ---------------------------------------------------------------------------
__global__ void rowstats_kernel(const float* __restrict__ sim,
                                float* __restrict__ rowmax,
                                float* __restrict__ rowsum) {
    __shared__ float sm[256];
    __shared__ float sv[256];
    const int t = threadIdx.x;
    const size_t row = blockIdx.x;            // 0 .. N*L-1
    const float* p = sim + row * SS;

    float m = -1e30f, s = 0.f;
    for (int c = t; c < SS; c += 256) {
        float x  = p[c];
        float nm = fmaxf(m, x);
        s = s * __expf(m - nm) + __expf(x - nm);
        m = nm;
    }
    sm[t] = m; sv[t] = s;
    __syncthreads();
    for (int w = 128; w > 0; w >>= 1) {
        if (t < w) {
            float m2 = sm[t + w], s2 = sv[t + w];
            float nm = fmaxf(sm[t], m2);
            sv[t] = sv[t] * __expf(sm[t] - nm) + s2 * __expf(m2 - nm);
            sm[t] = nm;
        }
        __syncthreads();
    }
    if (t == 0) { rowmax[row] = sm[0]; rowsum[row] = sv[0]; }
}

// ---------------------------------------------------------------------------
// 3b) Column stats: coalesced row-segment partials + combine (no atomics).
// ---------------------------------------------------------------------------
__global__ void colstats_partial(const float* __restrict__ sim,
                                 float* __restrict__ pm,
                                 float* __restrict__ ps) {
    const int c = blockIdx.x * 256 + threadIdx.x;
    if (c >= SS) return;
    const int seg = blockIdx.y;
    const int n   = blockIdx.z;
    const float* base = sim + ((size_t)n * LL + (size_t)seg * SEGROWS) * SS + c;

    float m = -1e30f, s = 0.f;
    for (int r = 0; r < SEGROWS; ++r) {
        float x  = base[(size_t)r * SS];
        float nm = fmaxf(m, x);
        s = s * __expf(m - nm) + __expf(x - nm);
        m = nm;
    }
    const size_t o = ((size_t)n * SEGS + seg) * SS + c;
    pm[o] = m; ps[o] = s;
}

__global__ void colstats_combine(const float* __restrict__ pm,
                                 const float* __restrict__ ps,
                                 float* __restrict__ colmax,
                                 float* __restrict__ colsum) {
    const int c = blockIdx.x * 256 + threadIdx.x;
    if (c >= SS) return;
    const int n = blockIdx.y;
    float m = -1e30f, s = 0.f;
    for (int seg = 0; seg < SEGS; ++seg) {
        const size_t o = ((size_t)n * SEGS + seg) * SS + c;
        float m2 = pm[o], s2 = ps[o];
        float nm = fmaxf(m, m2);
        s = s * __expf(m - nm) + s2 * __expf(m2 - nm);
        m = nm;
    }
    colmax[(size_t)n * SS + c] = m;
    colsum[(size_t)n * SS + c] = s;
}

// ---------------------------------------------------------------------------
// 4) Fused final pass: conf01, conf10, mutual-NN + threshold + border mask,
//    mconf. One read of sim, four output writes.
// ---------------------------------------------------------------------------
__global__ void final_kernel(const float* __restrict__ sim,
                             const float* __restrict__ rowmax,
                             const float* __restrict__ rowsum,
                             const float* __restrict__ colmax,
                             const float* __restrict__ colsum,
                             float* __restrict__ out) {
    const int c = blockIdx.x * 256 + threadIdx.x;
    if (c >= SS) return;
    const int l = blockIdx.y;
    const int n = blockIdx.z;

    const size_t NLS = (size_t)NB * LL * SS;
    const size_t idx = ((size_t)n * LL + l) * SS + c;

    const float x  = sim[idx];
    const float rm = rowmax[(size_t)n * LL + l];
    const float rs = rowsum[(size_t)n * LL + l];
    const float cm = colmax[(size_t)n * SS + c];
    const float cs = colsum[(size_t)n * SS + c];

    const float c01 = __expf(x - rm) / rs;
    const float c10 = __expf(x - cm) / cs;
    const bool m01 = (c01 > THRV) && (x == rm);
    const bool m10 = (c10 > THRV) && (x == cm);
    const bool msk = (m01 || m10) && border_ok(l) && border_ok(c);

    out[idx]           = c01;
    out[NLS + idx]     = c10;
    out[2 * NLS + idx] = msk ? 1.0f : 0.0f;
    out[3 * NLS + idx] = msk ? fmaxf(c01, c10) : 0.0f;
}

// ---------------------------------------------------------------------------
extern "C" void kernel_launch(void* const* d_in, const int* in_sizes, int n_in,
                              void* d_out, int out_size, void* d_ws, size_t ws_size,
                              hipStream_t stream) {
    (void)in_sizes; (void)n_in; (void)out_size; (void)ws_size;
    const float* feat0 = (const float*)d_in[0];
    const float* feat1 = (const float*)d_in[1];
    const float* Wm    = (const float*)d_in[2];
    const float* bias  = (const float*)d_in[3];
    // d_in[4], d_in[5] are all-true masks -> unused.

    char*  ws  = (char*)d_ws;
    size_t off = 0;
    auto alloc = [&](size_t bytes) -> void* {
        void* p = ws + off;
        off = (off + bytes + 255) & ~(size_t)255;
        return p;
    };

    unsigned short* fb   = (unsigned short*)alloc((size_t)TOTROWS * DD * sizeof(unsigned short));
    float* sim    = (float*)alloc((size_t)NB * LL * SS * sizeof(float));
    float* rowmax = (float*)alloc((size_t)NB * LL * sizeof(float));
    float* rowsum = (float*)alloc((size_t)NB * LL * sizeof(float));
    float* colmax = (float*)alloc((size_t)NB * SS * sizeof(float));
    float* colsum = (float*)alloc((size_t)NB * SS * sizeof(float));
    float* pm     = (float*)alloc((size_t)NB * SEGS * SS * sizeof(float));
    float* ps     = (float*)alloc((size_t)NB * SEGS * SS * sizeof(float));

    const int CB = (SS + 255) / 256;  // 19 column blocks

    proj_kernel<<<TOTROWS / 64, 256, 0, stream>>>(feat0, feat1, Wm, bias, fb);
    sim_gemm_kernel<<<dim3(SS / 64, LL / 64, NB), 128, 0, stream>>>(fb, sim);
    rowstats_kernel<<<NLROWS, 256, 0, stream>>>(sim, rowmax, rowsum);
    colstats_partial<<<dim3(CB, SEGS, NB), 256, 0, stream>>>(sim, pm, ps);
    colstats_combine<<<dim3(CB, NB), 256, 0, stream>>>(pm, ps, colmax, colsum);
    final_kernel<<<dim3(CB, LL, NB), 256, 0, stream>>>(sim, rowmax, rowsum,
                                                       colmax, colsum, (float*)d_out);
}
